// BRITSModel_88373247082606
// MI455X (gfx1250) — matrix operand order repaired and checked by reading, not compiled, not run on hardware
//
#include <hip/hip_runtime.h>
#include <hip/hip_bf16.h>

#define T_LEN 65536
#define H 108
#define G4H 432            // 4*H gates; row 432 additionally carries reg_w
#define NW  7              // waves per block
#define BLK (NW * 32)      // 224 threads

typedef __attribute__((ext_vector_type(16))) _Float16 v16h;
typedef __attribute__((ext_vector_type(8)))  float    v8f;

__device__ __forceinline__ float sigmoid_fast(float x) {
    return 1.0f / (1.0f + __expf(-x));
}
__device__ __forceinline__ float tanh_fast(float x) {
    float e = __expf(-2.0f * x);
    return (1.0f - e) / (1.0f + e);
}

// One block per direction. Sequential scan over T; per step the augmented
// 433x108 matvec [W_hh; reg_w] @ h_dec runs on v_wmma_f32_16x16x32_f16 with
// all weights resident in VGPRs (loaded once before the loop). Row 432 of
// the WMMA output is reg_w . h_dec, i.e. the regression estimate, so no
// separate reduction phase (and only 2 barriers per step) is needed.
__global__ __launch_bounds__(BLK, 1)
void brits_rnn_kernel(
    const float* __restrict__ values,   const float* __restrict__ masks,
    const float* __restrict__ deltas_f, const float* __restrict__ deltas_b,
    const float* __restrict__ f_td_w,   const float* __restrict__ f_td_b,
    const float* __restrict__ f_reg_w,  const float* __restrict__ f_reg_b,
    const float* __restrict__ f_W_ih,   const float* __restrict__ f_W_hh,
    const float* __restrict__ f_b_ih,   const float* __restrict__ f_b_hh,
    const float* __restrict__ b_td_w,   const float* __restrict__ b_td_b,
    const float* __restrict__ b_reg_w,  const float* __restrict__ b_reg_b,
    const float* __restrict__ b_W_ih,   const float* __restrict__ b_W_hh,
    const float* __restrict__ b_b_ih,   const float* __restrict__ b_b_hh,
    float* __restrict__ ws_imp,         // [2*T] imputations (natural time order)
    float* __restrict__ ws_loss)        // [2]   per-direction loss
{
    const int dir  = blockIdx.x;        // 0 = forward, 1 = backward
    const int tid  = threadIdx.x;
    const int lane = tid & 31;
    const int wv   = tid >> 5;          // 0..6

    const float* deltas = dir ? deltas_b : deltas_f;
    const float* td_w   = dir ? b_td_w  : f_td_w;
    const float* td_b   = dir ? b_td_b  : f_td_b;
    const float* reg_w  = dir ? b_reg_w : f_reg_w;
    const float* reg_b  = dir ? b_reg_b : f_reg_b;
    const float* W_ih   = dir ? b_W_ih  : f_W_ih;
    const float* W_hh   = dir ? b_W_hh  : f_W_hh;
    const float* b_ih   = dir ? b_b_ih  : f_b_ih;
    const float* b_hh   = dir ? b_b_hh  : f_b_hh;

    __shared__ alignas(64) _Float16 hdech[128];     // decayed h, f16, zero-padded
    __shared__ alignas(64) _Float16 hzero[128];     // 256B of zeros for idle B lanes
    __shared__ alignas(64) float    gates_mm[456];  // [W_hh;reg_w]@h (448 used)

    // ---- Load [W_hh; reg_w] into resident WMMA A-tiles (f16), once. -------
    // A-tile layout (16x32 f16, contiguous K order, same as B): lane m<16 holds row Mbase+m,
// halfs j=0..15 -> K=j; lanes 16..31 hold the same
// rows with K offset +16. Row 432 = reg_w (regression dot).
    v16h Areg[4][4];
#pragma unroll
    for (int mt = 0; mt < 4; ++mt) {
        const int Mbase = (wv * 4 + mt) * 16;
        const int row   = Mbase + (lane & 15);
        const int koff  = (lane < 16) ? 0 : 16;
#pragma unroll
        for (int kt = 0; kt < 4; ++kt) {
            v16h a = {};
#pragma unroll
            for (int j = 0; j < 16; ++j) {
                const int k = kt * 32 + koff + j;
                float w = 0.0f;
                if (k < H) {
                    if (row < G4H)       w = W_hh[row * H + k];
                    else if (row == G4H) w = reg_w[k];
                }
                a[j] = (_Float16)w;
            }
            Areg[mt][kt] = a;
        }
    }

    // ---- Per-element parameters held in registers (thread i owns state i) --
    float tdw = 0.f, tdb = 0.f;
    float wih0[4] = {0.f,0.f,0.f,0.f};
    float wih1[4] = {0.f,0.f,0.f,0.f};
    float bsum[4] = {0.f,0.f,0.f,0.f};
    if (tid < H) {
        tdw = td_w[tid]; tdb = td_b[tid];
#pragma unroll
        for (int q = 0; q < 4; ++q) {
            const int g = tid + q * H;
            wih0[q] = W_ih[2 * g];
            wih1[q] = W_ih[2 * g + 1];
            bsum[q] = b_ih[g] + b_hh[g];
        }
    }
    const float regb = reg_b[0];

    // zero pads / zero block once (never rewritten inside the loop)
    if (tid < 128) hzero[tid] = (_Float16)0.0f;
    if (tid >= H && tid < 128) hdech[tid] = (_Float16)0.0f;
    __syncthreads();

    // Per-lane loop-invariant B base: lane 0 reads K=0..15 of each chunk,
    // lane 16 reads K=16..31, every other lane streams zeros. No masking
    // VALU work needed per step.
    const _Float16* bbase = (lane == 0)  ? &hdech[0]
                          : (lane == 16) ? &hdech[16]
                                         : &hzero[0];
    // D column-0 writers: lanes 0 (rows +0..7) and 16 (rows +8..15) only.
    const bool storer = (lane & 15) == 0;
    const int  half8  = (lane >> 4) << 3;

    float h = 0.0f, c = 0.0f, loss = 0.0f;

    for (int t = 0; t < T_LEN; ++t) {
        const int pos = dir ? (T_LEN - 1 - t) : t;

        // ---- Phase A: temporal decay --------------------------------------
        if (tid < H) {
            const float dlt   = deltas[t];
            const float gamma = __expf(-fmaxf(fmaf(dlt, tdw, tdb), 0.0f));
            hdech[tid] = (_Float16)(h * gamma);
        }
        __syncthreads();

        // ---- Phase B: gates_mm = [W_hh; reg_w] @ h_dec via WMMA -----------
        v16h B[4];
#pragma unroll
        for (int kt = 0; kt < 4; ++kt)
            B[kt] = *(const v16h*)&bbase[kt * 32];

        v8f accs[4];                    // 4 independent accumulation chains
#pragma unroll
        for (int mt = 0; mt < 4; ++mt) {
            v8f acc = {};
#pragma unroll
            for (int kt = 0; kt < 4; ++kt) {
                acc = __builtin_amdgcn_wmma_f32_16x16x32_f16(
                    /*neg_a=*/false, Areg[mt][kt],
                    /*neg_b=*/false, B[kt],
                    /*c_mod=*/(short)0, acc,
                    /*reuse_a=*/false, /*reuse_b=*/false);
            }
            accs[mt] = acc;
        }
        // Single predicated store batch (one branch per step).
        if (storer) {
#pragma unroll
            for (int mt = 0; mt < 4; ++mt)
                *(v8f*)&gates_mm[(wv * 4 + mt) * 16 + half8] = accs[mt];
        }
        __syncthreads();

        // ---- Phase C: regression + LSTM pointwise update ------------------
        if (tid < H) {
            const float x   = values[pos];
            const float m   = masks[pos];
            const float x_h = gates_mm[G4H] + regb;     // reg_w . h_dec + reg_b
            const float xc  = fmaf(m, x, (1.0f - m) * x_h);

            const float gi = gates_mm[tid]         + fmaf(xc, wih0[0], fmaf(m, wih1[0], bsum[0]));
            const float gf = gates_mm[tid + H]     + fmaf(xc, wih0[1], fmaf(m, wih1[1], bsum[1]));
            const float gg = gates_mm[tid + 2 * H] + fmaf(xc, wih0[2], fmaf(m, wih1[2], bsum[2]));
            const float go = gates_mm[tid + 3 * H] + fmaf(xc, wih0[3], fmaf(m, wih1[3], bsum[3]));
            const float ig = sigmoid_fast(gi);
            const float fg = sigmoid_fast(gf);
            const float cg = tanh_fast(gg);
            const float og = sigmoid_fast(go);
            c = fmaf(fg, c, ig * cg);
            h = og * tanh_fast(c);

            if (tid == 0) {
                loss += fabsf(x - x_h) * m / (m + 1e-5f);
                ws_imp[dir * T_LEN + pos] = xc;   // backward stored re-reversed
            }
        }
        // barrier at top of next iteration separates phase C's gates_mm
        // reads from the next step's WMMA stores
    }

    if (tid == 0) ws_loss[dir] = loss;
}

// out[1+t] = 0.5*(imp_f+imp_b); per-block partial sums of |imp_f-imp_b|
__global__ void brits_combine_kernel(const float* __restrict__ ws_imp,
                                     float* __restrict__ out,
                                     float* __restrict__ ws_part)
{
    __shared__ float sh[256];
    const int t = blockIdx.x * 256 + threadIdx.x;
    const float f = ws_imp[t];
    const float b = ws_imp[T_LEN + t];
    out[1 + t] = 0.5f * (f + b);
    sh[threadIdx.x] = fabsf(f - b);
    __syncthreads();
#pragma unroll
    for (int s = 128; s > 0; s >>= 1) {
        if (threadIdx.x < s) sh[threadIdx.x] += sh[threadIdx.x + s];
        __syncthreads();
    }
    if (threadIdx.x == 0) ws_part[blockIdx.x] = sh[0];
}

__global__ void brits_finalize_kernel(const float* __restrict__ ws_part,
                                      const float* __restrict__ ws_loss,
                                      float* __restrict__ out)
{
    __shared__ float sh[256];
    sh[threadIdx.x] = ws_part[threadIdx.x];
    __syncthreads();
#pragma unroll
    for (int s = 128; s > 0; s >>= 1) {
        if (threadIdx.x < s) sh[threadIdx.x] += sh[threadIdx.x + s];
        __syncthreads();
    }
    if (threadIdx.x == 0)
        out[0] = 0.3f * (ws_loss[0] + ws_loss[1]) + sh[0] * (1.0f / (float)T_LEN);
}

extern "C" void kernel_launch(void* const* d_in, const int* in_sizes, int n_in,
                              void* d_out, int out_size, void* d_ws, size_t ws_size,
                              hipStream_t stream)
{
    (void)in_sizes; (void)n_in; (void)out_size; (void)ws_size;

    const float* values   = (const float*)d_in[0];
    const float* masks    = (const float*)d_in[1];
    const float* deltas_f = (const float*)d_in[2];
    const float* deltas_b = (const float*)d_in[3];
    const float* f_td_w   = (const float*)d_in[4];
    const float* f_td_b   = (const float*)d_in[5];
    const float* f_reg_w  = (const float*)d_in[6];
    const float* f_reg_b  = (const float*)d_in[7];
    const float* f_W_ih   = (const float*)d_in[8];
    const float* f_W_hh   = (const float*)d_in[9];
    const float* f_b_ih   = (const float*)d_in[10];
    const float* f_b_hh   = (const float*)d_in[11];
    const float* b_td_w   = (const float*)d_in[12];
    const float* b_td_b   = (const float*)d_in[13];
    const float* b_reg_w  = (const float*)d_in[14];
    const float* b_reg_b  = (const float*)d_in[15];
    const float* b_W_ih   = (const float*)d_in[16];
    const float* b_W_hh   = (const float*)d_in[17];
    const float* b_b_ih   = (const float*)d_in[18];
    const float* b_b_hh   = (const float*)d_in[19];

    float* out     = (float*)d_out;
    float* ws      = (float*)d_ws;
    float* ws_imp  = ws;                    // 2*T floats
    float* ws_loss = ws + 2 * T_LEN;        // 2 floats
    float* ws_part = ws + 2 * T_LEN + 2;    // 256 floats

    brits_rnn_kernel<<<2, BLK, 0, stream>>>(
        values, masks, deltas_f, deltas_b,
        f_td_w, f_td_b, f_reg_w, f_reg_b, f_W_ih, f_W_hh, f_b_ih, f_b_hh,
        b_td_w, b_td_b, b_reg_w, b_reg_b, b_W_ih, b_W_hh, b_b_ih, b_b_hh,
        ws_imp, ws_loss);

    brits_combine_kernel<<<T_LEN / 256, 256, 0, stream>>>(ws_imp, out, ws_part);
    brits_finalize_kernel<<<1, 256, 0, stream>>>(ws_part, ws_loss, out);
}
